// WildIDClassifier_42107859370068
// MI455X (gfx1250) — compile-verified, matrix operand order checked by reading, not run on hardware
//
#include <hip/hip_runtime.h>

// ---------------------------------------------------------------------------
// CDNA5 (gfx1250) WMMA types + helpers
// ---------------------------------------------------------------------------
typedef __attribute__((ext_vector_type(16))) _Float16 v16h;
typedef __attribute__((ext_vector_type(8)))  _Float16 v8h;
typedef __attribute__((ext_vector_type(8)))  float    v8f;

static __device__ __forceinline__ v8f zero8() {
  v8f z;
#pragma unroll
  for (int i = 0; i < 8; ++i) z[i] = 0.0f;
  return z;
}

static __device__ __forceinline__ v8h zero8h() {
  v8h z;
#pragma unroll
  for (int i = 0; i < 8; ++i) z[i] = (_Float16)0.0f;
  return z;
}

static __device__ __forceinline__ v16h pack16(v8h lo, v8h hi) {
  v16h r;
#pragma unroll
  for (int i = 0; i < 8; ++i) { r[i] = lo[i]; r[i + 8] = hi[i]; }
  return r;
}

// D = A(16x32 f16) * B(32x16 f16) + C(16x16 f32)
static __device__ __forceinline__ v8f wmma_f16f32(v16h a, v16h b, v8f c) {
  return __builtin_amdgcn_wmma_f32_16x16x32_f16(false, a, false, b, (short)0, c,
                                                false, false);
}

static __device__ __forceinline__ float sigmoidf_(float x) {
  return 1.0f / (1.0f + expf(-x));
}

// Padded NHWC geometry: f padded +2 each side, t padded +2 each side (T=512)
#define TPAD 516

// ---------------------------------------------------------------------------
// Vectorized f16 zero-fill (halo init for padded NHWC buffers)
// ---------------------------------------------------------------------------
__global__ void zero_f16(_Float16* __restrict__ p, int n8) {
  int i = blockIdx.x * blockDim.x + threadIdx.x;
  if (i < n8) ((v8h*)p)[i] = zero8h();
}

// ---------------------------------------------------------------------------
// f32 -> f16 elementwise (LSTM weights)
// ---------------------------------------------------------------------------
__global__ void cvt_f32_f16(const float* __restrict__ s, _Float16* __restrict__ d, int n) {
  int i = blockIdx.x * blockDim.x + threadIdx.x;
  if (i < n) d[i] = (_Float16)s[i];
}

// ---------------------------------------------------------------------------
// Conv weight repack: (OC,IC,5,5) f32 -> (25, OC, IC) f16 (tap-major, so the
// WMMA B-fragment's K (=input channel) axis is contiguous).
// ---------------------------------------------------------------------------
__global__ void repack_conv_w(const float* __restrict__ src, _Float16* __restrict__ dst,
                              int OC, int IC) {
  int idx = blockIdx.x * blockDim.x + threadIdx.x;
  if (idx >= OC * IC * 25) return;
  const int c   = idx % IC;
  const int oc  = (idx / IC) % OC;
  const int tap = idx / (IC * OC);
  dst[idx] = (_Float16)src[((size_t)oc * IC + c) * 25 + tap];
}

// ---------------------------------------------------------------------------
// PCEN: per-(b,f) row IIR scan over T=512.  Output (16,128,512) f16
// (unpadded; conv1 uses a scalar fill path anyway).
// ---------------------------------------------------------------------------
__global__ void pcen_kernel(const float* __restrict__ x, _Float16* __restrict__ out,
                            const float* __restrict__ la, const float* __restrict__ ld,
                            const float* __restrict__ lr) {
  int row = blockIdx.x * blockDim.x + threadIdx.x;
  if (row >= 16 * 128) return;
  const float alpha = expf(la[0]) * 0.98f;
  const float delta = expf(ld[0]) * 2.0f;
  const float r     = expf(lr[0]) * 0.5f;
  const float dr    = powf(delta, r);
  const float* xr = x + (size_t)row * 512;
  _Float16* orow  = out + (size_t)row * 512;
  float m = 0.0f;
  for (int t = 0; t < 512; ++t) {
    float v = xr[t];
    m = 0.975f * m + 0.025f * v;
    float o = powf(v / powf(1e-6f + m, alpha) + delta, r) - dr;
    orow[t] = (_Float16)o;
  }
}

// ---------------------------------------------------------------------------
// Conv1 (IC=1): K = 25 taps packed into one 32-wide WMMA chunk.
// in: (B,128,512) f16; w: (25,64) f16; out padded NHWC (B,68,TPAD,64),
// interior at (+2,+2).  Grid (32, 1, B*64), block 128 (4 waves).
// ---------------------------------------------------------------------------
__global__ void conv1_pool_kernel(const _Float16* __restrict__ in,
                                  const _Float16* __restrict__ wp,
                                  const float* __restrict__ bias,
                                  const float* __restrict__ gamma,
                                  const float* __restrict__ beta,
                                  const float* __restrict__ mean,
                                  const float* __restrict__ var,
                                  _Float16* __restrict__ out) {
  const int T = 512, Fin = 128, OC = 64, Fp = 64, FpadOut = 68;
  const int lane = threadIdx.x & 31;
  const int wv   = threadIdx.x >> 5;
  const int t0   = blockIdx.x * 16;
  const int oc0  = wv * 16;
  const int b    = blockIdx.z / Fp;
  const int fp   = blockIdx.z % Fp;
  const int kb   = (lane < 16) ? 0 : 8;
  const int mA   = lane & 15;
  const int nB   = lane & 15;

  v16h a0, a1, bm;
#pragma unroll
  for (int i = 0; i < 16; ++i) {
    const int j = (i < 8) ? (kb + i) : (16 + kb + (i - 8));
    const bool jok = j < 25;
    const int kf = j / 5 - 2;
    const int kt = j % 5 - 2;
    const int f0 = 2 * fp + kf;
    const int tA = t0 + mA + kt;
    const bool tok = (tA >= 0) && (tA < T);
    bm[i] = jok ? wp[j * OC + oc0 + nB] : (_Float16)0.0f;
    a0[i] = (jok && tok && f0 >= 0 && f0 < Fin)
              ? in[(size_t)(b * Fin + f0) * T + tA] : (_Float16)0.0f;
    a1[i] = (jok && tok && (f0 + 1) >= 0 && (f0 + 1) < Fin)
              ? in[(size_t)(b * Fin + f0 + 1) * T + tA] : (_Float16)0.0f;
  }
  v8f acc0 = wmma_f16f32(a0, bm, zero8());
  v8f acc1 = wmma_f16f32(a1, bm, zero8());

  const int oc = oc0 + nB;
  const float sc = gamma[oc] * rsqrtf(var[oc] + 1e-5f);
  const float bi = bias[oc], mu = mean[oc], be = beta[oc];
#pragma unroll
  for (int e = 0; e < 8; ++e) {
    const int t = t0 + e + ((lane >> 4) << 3);
    float v0 = (acc0[e] + bi - mu) * sc + be; v0 = fmaxf(v0, 0.0f);
    float v1 = (acc1[e] + bi - mu) * sc + be; v1 = fmaxf(v1, 0.0f);
    out[((size_t)(b * FpadOut + fp + 2) * TPAD + t + 2) * OC + oc] =
        (_Float16)fmaxf(v0, v1);
  }
}

// ---------------------------------------------------------------------------
// Padded-NHWC implicit-GEMM conv (IC in {64,128}) + bias+BN+ReLU+freq-maxpool.
// in:  (B, Fin+4, TPAD, IC) f16, zero halo, interior at (+2,+2)
// w:   (25, OC, IC) f16
// out: (B, Fin/2+4, TPAD, OC) f16, interior at (+2,+2)
// No bounds checks anywhere: every fragment fill is unconditional pairs of
// 16B vector loads.  Each wave computes a 32(t) x 16(oc) pooled tile:
// per 32-wide K chunk -> 5 x 32B loads/lane, 4 WMMAs (B shared by 2 t-tiles,
// A pointers shared by the 2 pooled freq rows).
// Block 128 (4 waves = 64 oc).  Grid (T/32, OC/64, B*Fp).
// ---------------------------------------------------------------------------
__global__ void conv_pool_nhwc_kernel(const _Float16* __restrict__ in,
                                      const _Float16* __restrict__ wp,
                                      const float* __restrict__ bias,
                                      const float* __restrict__ gamma,
                                      const float* __restrict__ beta,
                                      const float* __restrict__ mean,
                                      const float* __restrict__ var,
                                      _Float16* __restrict__ out,
                                      int IC, int OC, int Fin) {
  const int FpadIn  = Fin + 4;
  const int Fp      = Fin >> 1;
  const int FpadOut = Fp + 4;
  const int lane = threadIdx.x & 31;
  const int wv   = threadIdx.x >> 5;
  const int t0   = blockIdx.x * 32;
  const int oc0  = blockIdx.y * 64 + wv * 16;
  const int b    = blockIdx.z / Fp;
  const int fp   = blockIdx.z % Fp;
  const int kb   = (lane < 16) ? 0 : 8;
  const int mA   = lane & 15;
  const int nB   = lane & 15;

  v8f acc00 = zero8(), acc01 = zero8();   // freq row 0, t-tiles 0/1
  v8f acc10 = zero8(), acc11 = zero8();   // freq row 1, t-tiles 0/1
  const size_t tt1 = (size_t)16 * IC;     // t-tile 1 offset

  for (int tap = 0; tap < 25; ++tap) {
    const int kf = tap / 5 - 2;
    const int kt = tap % 5 - 2;
    // padded coords: f -> 2fp+kf+2 (always in [0,Fin+3]), t -> t0+mA+kt+2
    const _Float16* base0 =
        in + ((size_t)(b * FpadIn + 2 * fp + kf + 2) * TPAD + (t0 + mA + kt + 2)) * IC;
    const _Float16* base1 = base0 + (size_t)TPAD * IC;
    const _Float16* wrow  = wp + (size_t)(tap * OC + oc0 + nB) * IC;
    for (int c0 = 0; c0 < IC; c0 += 32) {
      v16h bm  = pack16(*(const v8h*)(wrow + c0 + kb),
                        *(const v8h*)(wrow + c0 + kb + 16));
      v16h a00 = pack16(*(const v8h*)(base0 + c0 + kb),
                        *(const v8h*)(base0 + c0 + kb + 16));
      v16h a10 = pack16(*(const v8h*)(base1 + c0 + kb),
                        *(const v8h*)(base1 + c0 + kb + 16));
      v16h a01 = pack16(*(const v8h*)(base0 + tt1 + c0 + kb),
                        *(const v8h*)(base0 + tt1 + c0 + kb + 16));
      v16h a11 = pack16(*(const v8h*)(base1 + tt1 + c0 + kb),
                        *(const v8h*)(base1 + tt1 + c0 + kb + 16));
      acc00 = wmma_f16f32(a00, bm, acc00);
      acc10 = wmma_f16f32(a10, bm, acc10);
      acc01 = wmma_f16f32(a01, bm, acc01);
      acc11 = wmma_f16f32(a11, bm, acc11);
    }
  }

  const int oc = oc0 + nB;
  const float sc = gamma[oc] * rsqrtf(var[oc] + 1e-5f);
  const float bi = bias[oc], mu = mean[oc], be = beta[oc];
  const size_t orow = (size_t)(b * FpadOut + fp + 2) * TPAD;
#pragma unroll
  for (int e = 0; e < 8; ++e) {
    const int t = t0 + e + ((lane >> 4) << 3);
    float v0 = (acc00[e] + bi - mu) * sc + be; v0 = fmaxf(v0, 0.0f);
    float v1 = (acc10[e] + bi - mu) * sc + be; v1 = fmaxf(v1, 0.0f);
    out[(orow + t + 2) * OC + oc] = (_Float16)fmaxf(v0, v1);
    float w0 = (acc01[e] + bi - mu) * sc + be; w0 = fmaxf(w0, 0.0f);
    float w1 = (acc11[e] + bi - mu) * sc + be; w1 = fmaxf(w1, 0.0f);
    out[(orow + t + 16 + 2) * OC + oc] = (_Float16)fmaxf(w0, w1);
  }
}

// ---------------------------------------------------------------------------
// Padded conv4 output (B=16, 12, TPAD, 128), interior (+2,+2) ->
// rows r = t*16+b, feature = c*8+fr  (== transpose(0,3,1,2).reshape of NCHW).
// ---------------------------------------------------------------------------
__global__ void reshape_kernel(const _Float16* __restrict__ in, _Float16* __restrict__ out) {
  int idx = blockIdx.x * blockDim.x + threadIdx.x;
  if (idx >= 16 * 8 * 512 * 128) return;
  const int c  = idx & 127;
  const int t  = (idx >> 7) & 511;
  const int fr = (idx >> 16) & 7;
  const int b  = idx >> 19;
  out[((size_t)(t * 16 + b) << 10) + c * 8 + fr] =
      in[((size_t)(b * 12 + fr + 2) * TPAD + t + 2) * 128 + c];
}

// ---------------------------------------------------------------------------
// LSTM input projection GEMM: pre[r,n] = sum_k X[r,k]*W[n,k] + bih[n]
// X: (8192, D) f16 (r = t*16+b), W: (512, D) f16, out f32 (8192,512).
// ---------------------------------------------------------------------------
__global__ void gemm_pre_kernel(const _Float16* __restrict__ X,
                                const _Float16* __restrict__ W,
                                const float* __restrict__ bias,
                                float* __restrict__ out, int D) {
  const int lane = threadIdx.x & 31;
  const int wv   = threadIdx.x >> 5;
  const int r0   = blockIdx.x * 16;
  const int n0   = blockIdx.y * 64 + wv * 16;
  const int kb   = (lane < 16) ? 0 : 8;
  const _Float16* xrow = X + (size_t)(r0 + (lane & 15)) * D;
  const _Float16* wrow = W + (size_t)(n0 + (lane & 15)) * D;
  v8f acc = zero8();
  for (int k0 = 0; k0 < D; k0 += 32) {
    if (k0 + 64 <= D) __builtin_prefetch(xrow + k0 + 64, 0, 0);
    v16h a = pack16(*(const v8h*)(xrow + k0 + kb), *(const v8h*)(xrow + k0 + kb + 16));
    v16h b = pack16(*(const v8h*)(wrow + k0 + kb), *(const v8h*)(wrow + k0 + kb + 16));
    acc = wmma_f16f32(a, b, acc);
  }
  const int n = n0 + (lane & 15);
#pragma unroll
  for (int e = 0; e < 8; ++e) {
    const int r = r0 + e + ((lane >> 4) << 3);
    out[(size_t)r * 512 + n] = acc[e] + bias[n];
  }
}

// ---------------------------------------------------------------------------
// LSTM recurrence (one direction): single persistent workgroup, 512 steps.
// h lives in LDS (f16); cell state c lives in VGPRs (wave w owns hidden slice
// [w*16, w*16+16), so all 4 gate tiles for an (m,h) pair land in one lane).
// ---------------------------------------------------------------------------
__global__ void lstm_rec_kernel(const float* __restrict__ pre,     // (512*16, 512) f32
                                const _Float16* __restrict__ whh,  // (512,128) f16
                                const float* __restrict__ bhh,     // (512)
                                _Float16* __restrict__ out16,      // optional
                                float* __restrict__ out32,         // optional
                                int out_stride, int out_off, int reverse) {
  __shared__ _Float16 h_lds[16 * 128];
  const int lane = threadIdx.x & 31;
  const int wv   = threadIdx.x >> 5;          // 0..7
  const int kb   = (lane < 16) ? 0 : 8;
  const int mA   = lane & 15;
  const int hidx = wv * 16 + (lane & 15);

  for (int i = threadIdx.x; i < 16 * 128; i += blockDim.x) h_lds[i] = (_Float16)0.0f;
  float c[8];
#pragma unroll
  for (int e = 0; e < 8; ++e) c[e] = 0.0f;
  __syncthreads();

  for (int s = 0; s < 512; ++s) {
    const int t = reverse ? (511 - s) : s;
    v8f acc[4];
#pragma unroll
    for (int g = 0; g < 4; ++g) acc[g] = zero8();

    for (int k0 = 0; k0 < 128; k0 += 32) {
      v16h a = pack16(*(const v8h*)(&h_lds[mA * 128 + k0 + kb]),
                      *(const v8h*)(&h_lds[mA * 128 + k0 + kb + 16]));
#pragma unroll
      for (int g = 0; g < 4; ++g) {
        const _Float16* wr = whh + (size_t)(g * 128 + hidx) * 128;
        v16h b = pack16(*(const v8h*)(wr + k0 + kb), *(const v8h*)(wr + k0 + kb + 16));
        acc[g] = wmma_f16f32(a, b, acc[g]);
      }
    }
    __syncthreads();  // all waves done reading h_lds

#pragma unroll
    for (int e = 0; e < 8; ++e) {
      const int m = e + ((lane >> 4) << 3);
      const size_t rz = (size_t)(t * 16 + m) * 512;
      const float iv = acc[0][e] + pre[rz +   0 + hidx] + bhh[  0 + hidx];
      const float fv = acc[1][e] + pre[rz + 128 + hidx] + bhh[128 + hidx];
      const float gv = acc[2][e] + pre[rz + 256 + hidx] + bhh[256 + hidx];
      const float ov = acc[3][e] + pre[rz + 384 + hidx] + bhh[384 + hidx];
      const float ci = sigmoidf_(fv) * c[e] + sigmoidf_(iv) * tanhf(gv);
      c[e] = ci;
      const float h = sigmoidf_(ov) * tanhf(ci);
      h_lds[m * 128 + hidx] = (_Float16)h;
      const size_t oi = (size_t)(t * 16 + m) * out_stride + out_off + hidx;
      if (out16) out16[oi] = (_Float16)h;
      if (out32) out32[oi] = h;
    }
    __syncthreads();  // h update visible before next step
  }
}

// ---------------------------------------------------------------------------
// Mean over T, then tiny FC (16x50)
// ---------------------------------------------------------------------------
__global__ void mean_kernel(const float* __restrict__ h, float* __restrict__ m) {
  int idx = blockIdx.x * blockDim.x + threadIdx.x;  // 16*256
  if (idx >= 4096) return;
  const int b = idx >> 8, k = idx & 255;
  float s = 0.0f;
  for (int t = 0; t < 512; ++t) s += h[((size_t)(t * 16 + b) << 8) + k];
  m[idx] = s * (1.0f / 512.0f);
}

__global__ void fc_kernel(const float* __restrict__ m, const float* __restrict__ w,
                          const float* __restrict__ bias, float* __restrict__ out) {
  int idx = blockIdx.x * blockDim.x + threadIdx.x;
  if (idx >= 16 * 50) return;
  const int b = idx / 50, o = idx % 50;
  float s = bias[o];
  for (int k = 0; k < 256; ++k) s += m[b * 256 + k] * w[o * 256 + k];
  out[idx] = s;
}

// ---------------------------------------------------------------------------
// Launch
// ---------------------------------------------------------------------------
extern "C" void kernel_launch(void* const* d_in, const int* in_sizes, int n_in,
                              void* d_out, int out_size, void* d_ws, size_t ws_size,
                              hipStream_t stream) {
  (void)in_sizes; (void)n_in; (void)out_size; (void)ws_size;
  const float* x = (const float*)d_in[0];
  const float* cw[4]  = {(const float*)d_in[1],  (const float*)d_in[7],  (const float*)d_in[13], (const float*)d_in[19]};
  const float* cb[4]  = {(const float*)d_in[2],  (const float*)d_in[8],  (const float*)d_in[14], (const float*)d_in[20]};
  const float* cg[4]  = {(const float*)d_in[3],  (const float*)d_in[9],  (const float*)d_in[15], (const float*)d_in[21]};
  const float* cbe[4] = {(const float*)d_in[4],  (const float*)d_in[10], (const float*)d_in[16], (const float*)d_in[22]};
  const float* cmu[4] = {(const float*)d_in[5],  (const float*)d_in[11], (const float*)d_in[17], (const float*)d_in[23]};
  const float* cv[4]  = {(const float*)d_in[6],  (const float*)d_in[12], (const float*)d_in[18], (const float*)d_in[24]};
  const float* fc_b = (const float*)d_in[25];
  const float* fc_w = (const float*)d_in[26];
  const float* la = (const float*)d_in[27];
  const float* ld = (const float*)d_in[28];
  const float* lr = (const float*)d_in[29];
  const float* wih0b = (const float*)d_in[30]; const float* whh0b = (const float*)d_in[31];
  const float* bih0b = (const float*)d_in[32]; const float* bhh0b = (const float*)d_in[33];
  const float* wih0f = (const float*)d_in[34]; const float* whh0f = (const float*)d_in[35];
  const float* bih0f = (const float*)d_in[36]; const float* bhh0f = (const float*)d_in[37];
  const float* wih1b = (const float*)d_in[38]; const float* whh1b = (const float*)d_in[39];
  const float* bih1b = (const float*)d_in[40]; const float* bhh1b = (const float*)d_in[41];
  const float* wih1f = (const float*)d_in[42]; const float* whh1f = (const float*)d_in[43];
  const float* bih1f = (const float*)d_in[44]; const float* bhh1f = (const float*)d_in[45];
  float* out = (float*)d_out;

  char* ws = (char*)d_ws;
  size_t off = 0;
  auto alloc = [&](size_t bytes) -> char* {
    char* p = ws + off; off += (bytes + 255) & ~(size_t)255; return p;
  };
  // padded NHWC ping-pong buffers; max stage = conv2 out (16,36,516,128)
  const size_t BUFH = 38043648ull;  // halves
  _Float16* pcen16 = (_Float16*)alloc(1048576ull * 2);   // (16,128,512)
  _Float16* bufA   = (_Float16*)alloc(BUFH * 2);
  _Float16* bufB   = (_Float16*)alloc(BUFH * 2);
  _Float16* x1     = (_Float16*)alloc(8388608ull * 2);   // (8192,1024)
  float*    pre_f  = (float*)alloc(4194304ull * 4);      // (8192,512)
  float*    pre_b  = (float*)alloc(4194304ull * 4);
  _Float16* hcat1  = (_Float16*)alloc(2097152ull * 2);   // (8192,256) f16
  float*    h2out  = (float*)alloc(2097152ull * 4);      // (8192,256) f32
  float*    meanb  = (float*)alloc(4096ull * 4);
  _Float16* wc[4]; const int wcn[4] = {1600, 204800, 409600, 409600};
  for (int i = 0; i < 4; ++i) wc[i] = (_Float16*)alloc((size_t)wcn[i] * 2);
  _Float16* wih0f16 = (_Float16*)alloc(524288ull * 2);
  _Float16* wih0b16 = (_Float16*)alloc(524288ull * 2);
  _Float16* wih1f16 = (_Float16*)alloc(131072ull * 2);
  _Float16* wih1b16 = (_Float16*)alloc(131072ull * 2);
  _Float16* whh0f16 = (_Float16*)alloc(65536ull * 2);
  _Float16* whh0b16 = (_Float16*)alloc(65536ull * 2);
  _Float16* whh1f16 = (_Float16*)alloc(65536ull * 2);
  _Float16* whh1b16 = (_Float16*)alloc(65536ull * 2);

  // weight repack/convert (re-run every call; deterministic)
  const int ocs[4] = {64, 128, 128, 128}, ics[4] = {1, 64, 128, 128};
  for (int i = 0; i < 4; ++i)
    repack_conv_w<<<(wcn[i] + 255) / 256, 256, 0, stream>>>(cw[i], wc[i], ocs[i], ics[i]);
  auto cvt = [&](const float* s, _Float16* d, int n) {
    cvt_f32_f16<<<(n + 255) / 256, 256, 0, stream>>>(s, d, n);
  };
  cvt(wih0f, wih0f16, 524288); cvt(wih0b, wih0b16, 524288);
  cvt(wih1f, wih1f16, 131072); cvt(wih1b, wih1b16, 131072);
  cvt(whh0f, whh0f16, 65536);  cvt(whh0b, whh0b16, 65536);
  cvt(whh1f, whh1f16, 65536);  cvt(whh1b, whh1b16, 65536);

  pcen_kernel<<<8, 256, 0, stream>>>(x, pcen16, la, ld, lr);

  const int nz = (int)(BUFH / 8);
  auto zero = [&](_Float16* p) { zero_f16<<<(nz + 255) / 256, 256, 0, stream>>>(p, nz); };

  // conv chain, padded NHWC end-to-end (halo zeroed before each stage write)
  zero(bufA);
  conv1_pool_kernel<<<dim3(32, 1, 16 * 64), 128, 0, stream>>>(
      pcen16, wc[0], cb[0], cg[0], cbe[0], cmu[0], cv[0], bufA);
  zero(bufB);
  conv_pool_nhwc_kernel<<<dim3(16, 2, 16 * 32), 128, 0, stream>>>(
      bufA, wc[1], cb[1], cg[1], cbe[1], cmu[1], cv[1], bufB,  64, 128, 64);
  zero(bufA);
  conv_pool_nhwc_kernel<<<dim3(16, 2, 16 * 16), 128, 0, stream>>>(
      bufB, wc[2], cb[2], cg[2], cbe[2], cmu[2], cv[2], bufA, 128, 128, 32);
  zero(bufB);
  conv_pool_nhwc_kernel<<<dim3(16, 2, 16 *  8), 128, 0, stream>>>(
      bufA, wc[3], cb[3], cg[3], cbe[3], cmu[3], cv[3], bufB, 128, 128, 16);

  reshape_kernel<<<32768, 256, 0, stream>>>(bufB, x1);

  // BiLSTM layer 1
  gemm_pre_kernel<<<dim3(512, 8), 128, 0, stream>>>(x1, wih0f16, bih0f, pre_f, 1024);
  gemm_pre_kernel<<<dim3(512, 8), 128, 0, stream>>>(x1, wih0b16, bih0b, pre_b, 1024);
  lstm_rec_kernel<<<1, 256, 0, stream>>>(pre_f, whh0f16, bhh0f, hcat1, (float*)nullptr, 256,   0, 0);
  lstm_rec_kernel<<<1, 256, 0, stream>>>(pre_b, whh0b16, bhh0b, hcat1, (float*)nullptr, 256, 128, 1);

  // BiLSTM layer 2
  gemm_pre_kernel<<<dim3(512, 8), 128, 0, stream>>>(hcat1, wih1f16, bih1f, pre_f, 256);
  gemm_pre_kernel<<<dim3(512, 8), 128, 0, stream>>>(hcat1, wih1b16, bih1b, pre_b, 256);
  lstm_rec_kernel<<<1, 256, 0, stream>>>(pre_f, whh1f16, bhh1f, (_Float16*)nullptr, h2out, 256,   0, 0);
  lstm_rec_kernel<<<1, 256, 0, stream>>>(pre_b, whh1b16, bhh1b, (_Float16*)nullptr, h2out, 256, 128, 1);

  mean_kernel<<<16, 256, 0, stream>>>(h2out, meanb);
  fc_kernel<<<4, 256, 0, stream>>>(meanb, fc_w, fc_b, out);
}
